// Block_47485158425016
// MI455X (gfx1250) — matrix operand order repaired and checked by reading, not compiled, not run on hardware
//
#include <hip/hip_runtime.h>

// ---------------------------------------------------------------------------
// Types for CDNA5 WMMA (wave32): bf16 16x16x32, f32 accumulate
// ---------------------------------------------------------------------------
typedef __attribute__((ext_vector_type(16))) __bf16 v16bf;
typedef __attribute__((ext_vector_type(8)))  __bf16 v8bf;
typedef __attribute__((ext_vector_type(8)))  float  v8f;

#define L_SEQ   1024
#define D_MODEL 2048
#define D_IN    4096
#define N_ST    16
#define DT_RANK 128
#define D_XDBL  160   // dt_rank + 2*n

#define WMMA_BF16(a, b, c) \
    __builtin_amdgcn_wmma_f32_16x16x32_bf16(false, (a), false, (b), (short)0, (c), false, false)

// Async global->LDS 16B copy (gfx1250, tracked by ASYNCcnt).
// ldsaddr: 32-bit LDS byte address; gptr: 64-bit global address.
#define ASYNC_G2L_B128(ldsaddr, gptr)                                    \
    asm volatile("global_load_async_to_lds_b128 %0, %1, off"             \
                 :: "v"(ldsaddr), "v"(gptr) : "memory")
#define WAIT_ASYNCCNT0() asm volatile("s_wait_asynccnt 0x0" ::: "memory")

// ---------------------------------------------------------------------------
// A-fragment combine: 16-bit A 16x32 layout (ISA 7.12.2):
//   lane<16 : elems 0..7 -> K=k0..k0+7,   elems 8..15 -> K=k0+16..k0+23
//   lane>=16: elems 0..7 -> K=k0+8..+15,  elems 8..15 -> K=k0+24..+31
// p points at row base + 8*(lane>>4)
// ---------------------------------------------------------------------------
static __device__ inline v16bf load_a_frag(const __bf16* p) {
    v8bf lo = *(const v8bf*)(p);
    v8bf hi = *(const v8bf*)(p + 16);
    return __builtin_shufflevector(lo, hi, 0,1,2,3,4,5,6,7,8,9,10,11,12,13,14,15);
}

// ---------------------------------------------------------------------------
// LDS-tiled GEMM: C[MxN] f32 = A[MxK] bf16 * W[NxK]^T bf16 (both row-major).
// Block = 256 threads (8 waves), block tile 128x128, K-step 32, double-
// buffered LDS panels filled by ASYNC global->LDS copies. Wave (2x4 grid)
// computes 64x32 = 8 WMMA frags/step. Requires M%128==0, N%128==0, K%32==0.
// ---------------------------------------------------------------------------
#define BM 128
#define BN 128
#define KS 32
#define LSTR 48   // padded LDS row stride in bf16 (96B: 32B-aligned, bank-staggered)

__global__ __launch_bounds__(256)
void wmma_gemm_lds_kernel(const __bf16* __restrict__ A,
                          const __bf16* __restrict__ W,
                          float* __restrict__ C,
                          int M, int N, int K)
{
    __shared__ __bf16 As[2][BM * LSTR];
    __shared__ __bf16 Bs[2][BN * LSTR];

    const int tid  = threadIdx.x;
    const int lane = tid & 31;
    const int wid  = tid >> 5;
    const int half = lane >> 4;
    const int l16  = lane & 15;
    const int mW   = (wid >> 2) << 6;   // wave row offset in block tile (0,64)
    const int nW   = (wid & 3) << 5;    // wave col offset in block tile (0..96)

    const int nB = N / BN;
    const int m0 = (blockIdx.x / nB) * BM;
    const int n0 = (blockIdx.x % nB) * BN;

    // Cooperative staging: each thread moves 2 x 16B chunks of A and of B per step.
    const int ldRow = tid >> 2;          // 0..63
    const int ldCol = (tid & 3) * 8;     // 0,8,16,24 (bf16 elems)
    const __bf16* Ag = A + (size_t)(m0 + ldRow) * K + ldCol;
    const __bf16* Bg = W + (size_t)(n0 + ldRow) * K + ldCol;
    const size_t rsA = (size_t)64 * K;   // +64 rows

    // LDS byte addresses for this thread's destinations (low 32 bits of
    // the generic pointer are the LDS aperture offset).
    const unsigned bufBytes = (unsigned)(BM * LSTR * 2);
    const unsigned aDst0 = (unsigned)(uintptr_t)(&As[0][ldRow * LSTR + ldCol]);
    const unsigned aDst1 = aDst0 + (unsigned)(64 * LSTR * 2);
    const unsigned bDst0 = (unsigned)(uintptr_t)(&Bs[0][ldRow * LSTR + ldCol]);
    const unsigned bDst1 = bDst0 + (unsigned)(64 * LSTR * 2);

    // Stage K-step panel (k elems offset) into LDS buffer `buf` (async).
    auto stage = [&](int buf, int kOff) {
        const unsigned bo = buf ? bufBytes : 0u;
        ASYNC_G2L_B128(aDst0 + bo, Ag + kOff);
        ASYNC_G2L_B128(aDst1 + bo, Ag + kOff + rsA);
        ASYNC_G2L_B128(bDst0 + bo, Bg + kOff);
        ASYNC_G2L_B128(bDst1 + bo, Bg + kOff + rsA);
    };

    // Preload K-step 0 into buffer 0
    stage(0, 0);
    WAIT_ASYNCCNT0();
    __syncthreads();

    v8f acc[4][2] = {};
    const int nK = K / KS;
    for (int ks = 0; ks < nK; ++ks) {
        const int cur = ks & 1;
        const bool more = (ks + 1 < nK);
        if (more) stage(cur ^ 1, (ks + 1) * KS);   // async fill of other buffer

        // Compute on current LDS buffer
        const __bf16* Asb = &As[cur][0];
        const __bf16* Bsb = &Bs[cur][0];
        v16bf bf0 = load_a_frag(Bsb + (nW + l16) * LSTR + 8 * half);
        v16bf bf1 = load_a_frag(Bsb + (nW + 16 + l16) * LSTR + 8 * half);
#pragma unroll
        for (int mt = 0; mt < 4; ++mt) {
            const __bf16* ap = Asb + (mW + mt * 16 + l16) * LSTR + 8 * half;
            v16bf af = load_a_frag(ap);
            acc[mt][0] = WMMA_BF16(af, bf0, acc[mt][0]);
            acc[mt][1] = WMMA_BF16(af, bf1, acc[mt][1]);
        }

        if (more) {                    // our async writes done, then block-wide swap
            WAIT_ASYNCCNT0();
            __syncthreads();
        }
    }

    // Store: VGPR r -> M = r + 8*half (per 16-row frag); N = l16 (+16 per nt)
    float* cr = C + (size_t)(m0 + mW + 8 * half) * N + (n0 + nW + l16);
#pragma unroll
    for (int mt = 0; mt < 4; ++mt) {
#pragma unroll
        for (int r = 0; r < 8; ++r) {
            const size_t ro = (size_t)(mt * 16 + r) * N;
            cr[ro]      = acc[mt][0][r];
            cr[ro + 16] = acc[mt][1][r];
        }
    }
}

// ---------------------------------------------------------------------------
// Direct-global WMMA GEMM (wave tile 32x32) — used for x_proj (N=160).
// ---------------------------------------------------------------------------
__global__ __launch_bounds__(256)
void wmma_gemm_bf16_kernel(const __bf16* __restrict__ A,
                           const __bf16* __restrict__ W,
                           float* __restrict__ C,
                           int M, int N, int K)
{
    const int lane = threadIdx.x & 31;
    const int wid  = threadIdx.x >> 5;
    const int nT   = N >> 5;
    const int mT   = M >> 5;
    const int tile = blockIdx.x * 8 + wid;
    if (tile >= mT * nT) return;            // wave-uniform: EXEC stays full

    const int m0   = (tile / nT) << 5;
    const int n0   = (tile % nT) << 5;
    const int half = lane >> 4;
    const int l16  = lane & 15;

    const __bf16* a0p = A + (size_t)(m0 + l16) * K + 8 * half;
    const __bf16* a1p = a0p + (size_t)16 * K;
    const __bf16* b0p = W + (size_t)(n0 + l16) * K + 8 * half;
    const __bf16* b1p = b0p + (size_t)16 * K;

    v8f c00 = {}, c01 = {}, c10 = {}, c11 = {};
    for (int k = 0; k < K; k += 32) {
        v16bf a0 = load_a_frag(a0p + k);
        v16bf a1 = load_a_frag(a1p + k);
        v16bf b0 = load_a_frag(b0p + k);
        v16bf b1 = load_a_frag(b1p + k);
        c00 = WMMA_BF16(a0, b0, c00);
        c01 = WMMA_BF16(a0, b1, c01);
        c10 = WMMA_BF16(a1, b0, c10);
        c11 = WMMA_BF16(a1, b1, c11);
    }

    float* cr = C + (size_t)(m0 + 8 * half) * N + n0 + l16;
#pragma unroll
    for (int r = 0; r < 8; ++r) {
        cr[(size_t)r * N]             = c00[r];
        cr[(size_t)r * N + 16]        = c01[r];
        cr[(size_t)(r + 16) * N]      = c10[r];
        cr[(size_t)(r + 16) * N + 16] = c11[r];
    }
}

// ---------------------------------------------------------------------------
// RMSNorm row kernel: one block per row of (1024 x 2048); writes bf16.
// ---------------------------------------------------------------------------
__global__ __launch_bounds__(256)
void rmsnorm_bf16_kernel(const float* __restrict__ x,
                         const float* __restrict__ w,
                         __bf16* __restrict__ out)
{
    __shared__ float sred[256];
    const int row = blockIdx.x;
    const float* xr = x + (size_t)row * D_MODEL;
    float ss = 0.f;
    for (int i = threadIdx.x; i < D_MODEL; i += 256) { float v = xr[i]; ss += v * v; }
    sred[threadIdx.x] = ss;
    __syncthreads();
    for (int s = 128; s > 0; s >>= 1) {
        if ((int)threadIdx.x < s) sred[threadIdx.x] += sred[threadIdx.x + s];
        __syncthreads();
    }
    const float scale = rsqrtf(sred[0] * (1.0f / D_MODEL) + 1e-5f);
    for (int i = threadIdx.x; i < D_MODEL; i += 256)
        out[(size_t)row * D_MODEL + i] = (__bf16)(xr[i] * scale * w[i]);
}

// f32 -> bf16 elementwise (weight staging)
__global__ void cvt_bf16_kernel(const float* __restrict__ in, __bf16* __restrict__ out, int n)
{
    int i = blockIdx.x * 256 + threadIdx.x;
    if (i < n) out[i] = (__bf16)in[i];
}

// Depthwise causal conv (d_conv=4, left pad 3) + bias + SiLU.
__global__ void conv_silu_kernel(const float* __restrict__ xz,
                                 const float* __restrict__ cw,
                                 const float* __restrict__ cb,
                                 float* __restrict__ xcf,
                                 __bf16* __restrict__ xcb)
{
    int i  = blockIdx.x * 256 + threadIdx.x;     // over L_SEQ * D_IN
    int l  = i >> 12;
    int ch = i & (D_IN - 1);
    float acc = cb[ch];
    const float* w = cw + ch * 4;
#pragma unroll
    for (int j = 0; j < 4; ++j) {
        int ll = l - 3 + j;
        if (ll >= 0) acc += xz[(size_t)ll * (2 * D_IN) + ch] * w[j];
    }
    float s = acc / (1.f + __expf(-acc));
    xcf[i] = s;
    xcb[i] = (__bf16)s;
}

// Extract delta part (first 128 cols of x_dbl) to bf16 for dt_proj GEMM.
__global__ void split_delta_kernel(const float* __restrict__ xdbl, __bf16* __restrict__ dbf)
{
    int i = blockIdx.x * 256 + threadIdx.x;      // over L_SEQ * DT_RANK
    int l = i >> 7;
    int r = i & (DT_RANK - 1);
    dbf[i] = (__bf16)xdbl[(size_t)l * D_XDBL + r];
}

// delta = softplus(dt_out + dt_proj_b), in place.
__global__ void softplus_kernel(float* __restrict__ dt, const float* __restrict__ bias)
{
    int i  = blockIdx.x * 256 + threadIdx.x;     // over L_SEQ * D_IN
    int ch = i & (D_IN - 1);
    float v = dt[i] + bias[ch];
    dt[i] = (v > 20.f) ? v : log1pf(__expf(v));
}

// Selective scan: one thread per channel d; 16 states in registers.
// Fuses +D*u, SiLU gate with res, and bf16 cast of y for out_proj.
__global__ __launch_bounds__(256)
void scan_kernel(const float* __restrict__ delta,
                 const float* __restrict__ xcf,
                 const float* __restrict__ xdbl,
                 const float* __restrict__ xz,
                 const float* __restrict__ A_log,
                 const float* __restrict__ Dvec,
                 __bf16* __restrict__ ybf)
{
    const int d = blockIdx.x * 256 + threadIdx.x;   // 0..4095
    float Arow[N_ST], h[N_ST];
#pragma unroll
    for (int n = 0; n < N_ST; ++n) {
        Arow[n] = -__expf(A_log[(size_t)d * N_ST + n]);
        h[n] = 0.f;
    }
    const float Dd = Dvec[d];
    for (int l = 0; l < L_SEQ; ++l) {
        const float dl = delta[(size_t)l * D_IN + d];
        const float u  = xcf[(size_t)l * D_IN + d];
        const float* Bp = xdbl + (size_t)l * D_XDBL + DT_RANK;
        const float* Cp = Bp + N_ST;
        const float du = dl * u;
        float y = u * Dd;
#pragma unroll
        for (int n = 0; n < N_ST; ++n) {
            h[n] = __expf(dl * Arow[n]) * h[n] + du * Bp[n];
            y += h[n] * Cp[n];
        }
        const float r = xz[(size_t)l * (2 * D_IN) + D_IN + d];
        y *= r / (1.f + __expf(-r));
        ybf[(size_t)l * D_IN + d] = (__bf16)y;
    }
}

// out = gemm_result + x (residual)
__global__ void resid_kernel(const float* __restrict__ mm, const float* __restrict__ x,
                             float* __restrict__ out)
{
    int i = blockIdx.x * 256 + threadIdx.x;      // over L_SEQ * D_MODEL
    out[i] = mm[i] + x[i];
}

// ---------------------------------------------------------------------------
// Host launcher
// ---------------------------------------------------------------------------
static inline int cdiv(long long a, long long b) { return (int)((a + b - 1) / b); }

extern "C" void kernel_launch(void* const* d_in, const int* in_sizes, int n_in,
                              void* d_out, int out_size, void* d_ws, size_t ws_size,
                              hipStream_t stream)
{
    const float* x        = (const float*)d_in[0];   // (1,1024,2048)
    const float* norm_w   = (const float*)d_in[1];   // (2048)
    const float* in_proj  = (const float*)d_in[2];   // (8192,2048)
    const float* conv_w   = (const float*)d_in[3];   // (4096,1,4)
    const float* conv_b   = (const float*)d_in[4];   // (4096)
    const float* x_proj   = (const float*)d_in[5];   // (160,4096)
    const float* dt_proj  = (const float*)d_in[6];   // (4096,128)
    const float* dt_b     = (const float*)d_in[7];   // (4096)
    const float* A_log    = (const float*)d_in[8];   // (4096,16)
    const float* Dvec     = (const float*)d_in[9];   // (4096)
    const float* out_proj = (const float*)d_in[10];  // (2048,4096)
    float* out = (float*)d_out;

    // Workspace carve-up (256B aligned)
    char*  ws  = (char*)d_ws;
    size_t off = 0;
    auto alloc = [&](size_t bytes) -> void* {
        void* p = ws + off;
        off += (bytes + 255) & ~(size_t)255;
        return p;
    };
    __bf16* in_proj_bf  = (__bf16*)alloc((size_t)2 * D_IN * D_MODEL * 2);   // 8192x2048
    __bf16* out_proj_bf = (__bf16*)alloc((size_t)D_MODEL * D_IN * 2);       // 2048x4096
    __bf16* x_proj_bf   = (__bf16*)alloc((size_t)D_XDBL * D_IN * 2);        // 160x4096
    __bf16* dt_proj_bf  = (__bf16*)alloc((size_t)D_IN * DT_RANK * 2);       // 4096x128
    __bf16* xn_bf       = (__bf16*)alloc((size_t)L_SEQ * D_MODEL * 2);
    __bf16* xc_bf       = (__bf16*)alloc((size_t)L_SEQ * D_IN * 2);
    __bf16* dIn_bf      = (__bf16*)alloc((size_t)L_SEQ * DT_RANK * 2);
    __bf16* y_bf        = (__bf16*)alloc((size_t)L_SEQ * D_IN * 2);
    float*  xz          = (float*)alloc((size_t)L_SEQ * 2 * D_IN * 4);      // 1024x8192
    float*  xc_f        = (float*)alloc((size_t)L_SEQ * D_IN * 4);
    float*  x_dbl       = (float*)alloc((size_t)L_SEQ * D_XDBL * 4);
    float*  delta       = (float*)alloc((size_t)L_SEQ * D_IN * 4);          // dt_out -> softplus in place
    float*  out_mm      = (float*)alloc((size_t)L_SEQ * D_MODEL * 4);
    (void)ws_size; (void)in_sizes; (void)n_in; (void)out_size;

    // 1) Stage weights as bf16
    {
        int n;
        n = 2 * D_IN * D_MODEL; cvt_bf16_kernel<<<cdiv(n,256),256,0,stream>>>(in_proj,  in_proj_bf,  n);
        n = D_MODEL * D_IN;     cvt_bf16_kernel<<<cdiv(n,256),256,0,stream>>>(out_proj, out_proj_bf, n);
        n = D_XDBL * D_IN;      cvt_bf16_kernel<<<cdiv(n,256),256,0,stream>>>(x_proj,   x_proj_bf,   n);
        n = D_IN * DT_RANK;     cvt_bf16_kernel<<<cdiv(n,256),256,0,stream>>>(dt_proj,  dt_proj_bf,  n);
    }

    // 2) RMSNorm -> bf16 activations
    rmsnorm_bf16_kernel<<<L_SEQ, 256, 0, stream>>>(x, norm_w, xn_bf);

    // 3) in_proj: xz = xn @ in_proj^T   (M=1024, N=8192, K=2048)  [LDS-tiled, async]
    wmma_gemm_lds_kernel<<<(L_SEQ/BM) * ((2*D_IN)/BN), 256, 0, stream>>>(
        xn_bf, in_proj_bf, xz, L_SEQ, 2*D_IN, D_MODEL);

    // 4) depthwise causal conv + SiLU
    conv_silu_kernel<<<cdiv((long long)L_SEQ*D_IN,256),256,0,stream>>>(xz, conv_w, conv_b, xc_f, xc_bf);

    // 5) x_proj: x_dbl = xc @ x_proj^T  (M=1024, N=160, K=4096)  [direct, N%128!=0]
    {
        int tiles = (L_SEQ/32) * (D_XDBL/32);
        wmma_gemm_bf16_kernel<<<cdiv(tiles,8),256,0,stream>>>(xc_bf, x_proj_bf, x_dbl,
                                                              L_SEQ, D_XDBL, D_IN);
    }

    // 6) delta slice -> bf16; dt_proj GEMM (M=1024,N=4096,K=128); softplus(+bias)
    split_delta_kernel<<<cdiv((long long)L_SEQ*DT_RANK,256),256,0,stream>>>(x_dbl, dIn_bf);
    wmma_gemm_lds_kernel<<<(L_SEQ/BM) * (D_IN/BN), 256, 0, stream>>>(
        dIn_bf, dt_proj_bf, delta, L_SEQ, D_IN, DT_RANK);
    softplus_kernel<<<cdiv((long long)L_SEQ*D_IN,256),256,0,stream>>>(delta, dt_b);

    // 7) selective scan + D skip + SiLU gate -> y (bf16)
    scan_kernel<<<D_IN/256, 256, 0, stream>>>(delta, xc_f, x_dbl, xz, A_log, Dvec, y_bf);

    // 8) out_proj: out_mm = y @ out_proj^T (M=1024, N=2048, K=4096), then +x
    wmma_gemm_lds_kernel<<<(L_SEQ/BM) * (D_MODEL/BN), 256, 0, stream>>>(
        y_bf, out_proj_bf, out_mm, L_SEQ, D_MODEL, D_IN);
    resid_kernel<<<cdiv((long long)L_SEQ*D_MODEL,256),256,0,stream>>>(out_mm, x, out);
}
